// SelfAttentionFA_90701119357485
// MI455X (gfx1250) — matrix operand order repaired and checked by reading, not compiled, not run on hardware
//
#include <hip/hip_runtime.h>
#include <hip/hip_bf16.h>

#define DIM    768
#define NHEADS 12
#define HD     64
#define BATCH  16
#define SEQ    1024
#define QKVOUT (3 * DIM)
#define MROWS  (BATCH * SEQ)   // 16384

typedef __attribute__((ext_vector_type(16))) _Float16 v16h;
typedef __attribute__((ext_vector_type(8)))  _Float16 v8h;
typedef __attribute__((ext_vector_type(8)))  float    v8f;

union AFrag { v16h v; v8h h[2]; };

static __device__ __forceinline__ v16h load_h16(const _Float16* p) {
  return *(const v16h*)p;
}
static __device__ __forceinline__ v8h load_h8(const _Float16* p) {
  return *(const v8h*)p;
}
static __device__ __forceinline__ v8f wmma16(v16h a, v16h b, v8f c) {
  // D(f32 16x16) = A(f16 16x32) * B(f16 32x16) + C
  return __builtin_amdgcn_wmma_f32_16x16x32_f16(false, a, false, b, (short)0, c,
                                                false, false);
}
static __device__ __forceinline__ v8f zero8() {
  v8f z = {0.f, 0.f, 0.f, 0.f, 0.f, 0.f, 0.f, 0.f};
  return z;
}

// ---------------------------------------------------------------- convert
__global__ void cvt_f32_f16_kernel(const float* __restrict__ src,
                                   _Float16* __restrict__ dst, int n) {
  int i = blockIdx.x * blockDim.x + threadIdx.x;
  if (i < n) dst[i] = (_Float16)src[i];
}

// ---------------------------------------------------------------- QKV GEMM
// C[16384, 2304] = x16[16384,768] @ qkv_w16[2304,768]^T + b ; scatter into
// q16/k16 as [B,H,N,hd] (q pre-scaled by hd^-0.5) and vT16 as [B,H,hd,N].
// Per wave: 32x64 output tile (2 A-frags share 4 B-frags -> 8 WMMA / chunk).
__global__ __launch_bounds__(256)
void qkv_gemm_kernel(const _Float16* __restrict__ x16,
                     const _Float16* __restrict__ w16,
                     const float* __restrict__ bias,
                     _Float16* __restrict__ q16,
                     _Float16* __restrict__ k16,
                     _Float16* __restrict__ vT16) {
  const int lane = threadIdx.x & 31;
  const int wave = threadIdx.x >> 5;
  const int l15 = lane & 15, hh = lane >> 4;
  const int row0 = (blockIdx.y * 8 + wave) * 32;   // 32-row M tile
  const int n0 = blockIdx.x * 64;                  // 64-col N tile

  v8f acc[2][4];
#pragma unroll
  for (int u = 0; u < 2; ++u)
#pragma unroll
    for (int t = 0; t < 4; ++t) acc[u][t] = zero8();

  for (int kc = 0; kc < DIM; kc += 32) {
    AFrag a0, a1;
    const _Float16* ap0 = x16 + (row0 + l15) * DIM + kc;
    const _Float16* ap1 = x16 + (row0 + 16 + l15) * DIM + kc;
    a0.h[0] = load_h8(ap0 + 16 * hh);
    a0.h[1] = load_h8(ap0 + 16 * hh + 8);
    a1.h[0] = load_h8(ap1 + 16 * hh);
    a1.h[1] = load_h8(ap1 + 16 * hh + 8);
#pragma unroll
    for (int t = 0; t < 4; ++t) {
      v16h b = load_h16(w16 + (n0 + t * 16 + l15) * DIM + kc + 16 * hh);
      acc[0][t] = wmma16(a0.v, b, acc[0][t]);
      acc[1][t] = wmma16(a1.v, b, acc[1][t]);
    }
  }

#pragma unroll
  for (int t = 0; t < 4; ++t) {
    const int col = n0 + t * 16 + l15;     // 0..2303
    const float bv = bias[col];
    const int s = col / DIM;               // 0=q 1=k 2=v
    const int cc = col - s * DIM;
    const int head = cc >> 6, d = cc & 63;
#pragma unroll
    for (int u = 0; u < 2; ++u) {
#pragma unroll
      for (int r = 0; r < 8; ++r) {
        const int row = row0 + u * 16 + r + 8 * hh;   // global token row
        const int bi = row >> 10, n = row & 1023;
        const float val = acc[u][t][r] + bv;
        if (s == 0)
          q16[((bi * NHEADS + head) * SEQ + n) * HD + d] = (_Float16)(val * 0.125f);
        else if (s == 1)
          k16[((bi * NHEADS + head) * SEQ + n) * HD + d] = (_Float16)val;
        else
          vT16[((bi * NHEADS + head) * HD + d) * SEQ + n] = (_Float16)val;
      }
    }
  }
}

// ---------------------------------------------------------------- attention
// Flash-attention: per wave = 32 query rows (2 tiles); compute S^T = K @ Q^T
// so P maps lane-locally onto the A-fragment of P@V (no cross-lane shuffles
// for the P re-layout). K A-frags and V B-frags are shared across both query
// tiles -> 16 WMMA per 16 b128-loads per 32-key step.
__global__ __launch_bounds__(256)
void attn_kernel(const _Float16* __restrict__ q16,
                 const _Float16* __restrict__ k16,
                 const _Float16* __restrict__ vT16,
                 _Float16* __restrict__ att16) {
  const int lane = threadIdx.x & 31;
  const int wave = threadIdx.x >> 5;
  const int l15 = lane & 15, hh = lane >> 4;
  const int bh = blockIdx.y;                       // 0..191
  const int b = bh / NHEADS, h = bh - b * NHEADS;
  const int q0 = blockIdx.x * 256 + wave * 32;     // 32-query tile base

  const _Float16* qp = q16 + (size_t)bh * SEQ * HD;
  const _Float16* kp = k16 + (size_t)bh * SEQ * HD;
  const _Float16* vp = vT16 + (size_t)bh * HD * SEQ;

  // Q^T as B fragments (column = query row, K contiguous), 2 query tiles.
  v16h qb[2][2];
#pragma unroll
  for (int u = 0; u < 2; ++u) {
    qb[u][0] = load_h16(qp + (q0 + u * 16 + l15) * HD + 16 * hh);
    qb[u][1] = load_h16(qp + (q0 + u * 16 + l15) * HD + 32 + 16 * hh);
  }

  float m[2] = {-__builtin_inff(), -__builtin_inff()};
  float l[2] = {0.f, 0.f};
  v8f acc[2][4];
#pragma unroll
  for (int u = 0; u < 2; ++u)
#pragma unroll
    for (int t = 0; t < 4; ++t) acc[u][t] = zero8();

  for (int j = 0; j < SEQ; j += 32) {
    // K rows as A fragments: two 16-key tiles, hd split in 2 K-chunks of 32.
    AFrag ka0l, ka0h, ka1l, ka1h;
    const _Float16* k0 = kp + (j + l15) * HD;
    const _Float16* k1 = kp + (j + 16 + l15) * HD;
    ka0l.h[0] = load_h8(k0 + 16 * hh);      ka0l.h[1] = load_h8(k0 + 8 + 16 * hh);
    ka0h.h[0] = load_h8(k0 + 32 + 16 * hh); ka0h.h[1] = load_h8(k0 + 40 + 16 * hh);
    ka1l.h[0] = load_h8(k1 + 16 * hh);      ka1l.h[1] = load_h8(k1 + 8 + 16 * hh);
    ka1h.h[0] = load_h8(k1 + 32 + 16 * hh); ka1h.h[1] = load_h8(k1 + 40 + 16 * hh);

    // prefetch next 32-key block of K and V (global_prefetch_b8)
    if (j + 32 < SEQ) {
      __builtin_prefetch(kp + (j + 32 + l15) * HD, 0, 3);
      __builtin_prefetch(kp + (j + 48 + l15) * HD, 0, 3);
      __builtin_prefetch(vp + (lane << 1) * SEQ + j + 32, 0, 3);
    }

    AFrag pA[2];
    float arow[2][8];
#pragma unroll
    for (int u = 0; u < 2; ++u) {
      // S^T tiles for query tile u (softmax scale folded into q16)
      v8f s0 = wmma16(ka0h.v, qb[u][1], wmma16(ka0l.v, qb[u][0], zero8()));
      v8f s1 = wmma16(ka1h.v, qb[u][1], wmma16(ka1l.v, qb[u][0], zero8()));

      // online softmax; lane owns query l15 (stats replicated across lane^16)
      float vmax = s0[0];
#pragma unroll
      for (int r = 0; r < 8; ++r) {
        vmax = fmaxf(vmax, s0[r]);
        vmax = fmaxf(vmax, s1[r]);
      }
      vmax = fmaxf(vmax, __shfl_xor(vmax, 16));
      const float m_new = fmaxf(m[u], vmax);
      const float alpha = __expf(m[u] - m_new);
      m[u] = m_new;

      float rs = 0.f;
#pragma unroll
      for (int r = 0; r < 8; ++r) {
        const float p0 = __expf(s0[r] - m_new);
        const float p1 = __expf(s1[r] - m_new);
        rs += p0 + p1;
        pA[u].v[r] = (_Float16)p0;       // keys 8*hh + r      (K 0..15 half)
        pA[u].v[8 + r] = (_Float16)p1;   // keys 16 + 8*hh + r (K 16..31 half)
      }
      rs += __shfl_xor(rs, 16);
      l[u] = l[u] * alpha + rs;

      // broadcast per-row rescale factors to accumulator layout (row=r+8*hh)
#pragma unroll
      for (int r = 0; r < 8; ++r) arow[u][r] = __shfl(alpha, r + 8 * hh);
    }

#pragma unroll
    for (int t = 0; t < 4; ++t) {
      AFrag vb;
      vb.h[0] = load_h8(vp + (t * 16 + l15) * SEQ + j + 8 * hh);
      vb.h[1] = load_h8(vp + (t * 16 + l15) * SEQ + j + 16 + 8 * hh);
#pragma unroll
      for (int u = 0; u < 2; ++u) {
#pragma unroll
        for (int r = 0; r < 8; ++r) acc[u][t][r] *= arow[u][r];
        acc[u][t] = wmma16(pA[u].v, vb.v, acc[u][t]);   // O[query,d] += P @ V
      }
    }
  }

#pragma unroll
  for (int u = 0; u < 2; ++u) {
    float lrow[8];
#pragma unroll
    for (int r = 0; r < 8; ++r) lrow[r] = __shfl(l[u], r + 8 * hh);
#pragma unroll
    for (int t = 0; t < 4; ++t) {
      const int col = h * HD + t * 16 + l15;
#pragma unroll
      for (int r = 0; r < 8; ++r) {
        const int n = q0 + u * 16 + r + 8 * hh;
        att16[((size_t)(b * SEQ + n)) * DIM + col] =
            (_Float16)(acc[u][t][r] / lrow[r]);
      }
    }
  }
}

// ---------------------------------------------------------------- proj GEMM
// out[16384,768] = att16 @ proj_w16^T + proj_b  (f32 output), 32x64 per wave.
__global__ __launch_bounds__(256)
void proj_gemm_kernel(const _Float16* __restrict__ a16,
                      const _Float16* __restrict__ w16,
                      const float* __restrict__ bias,
                      float* __restrict__ out) {
  const int lane = threadIdx.x & 31;
  const int wave = threadIdx.x >> 5;
  const int l15 = lane & 15, hh = lane >> 4;
  const int row0 = (blockIdx.y * 8 + wave) * 32;
  const int n0 = blockIdx.x * 64;

  v8f acc[2][4];
#pragma unroll
  for (int u = 0; u < 2; ++u)
#pragma unroll
    for (int t = 0; t < 4; ++t) acc[u][t] = zero8();

  for (int kc = 0; kc < DIM; kc += 32) {
    AFrag a0, a1;
    const _Float16* ap0 = a16 + (row0 + l15) * DIM + kc;
    const _Float16* ap1 = a16 + (row0 + 16 + l15) * DIM + kc;
    a0.h[0] = load_h8(ap0 + 16 * hh);
    a0.h[1] = load_h8(ap0 + 16 * hh + 8);
    a1.h[0] = load_h8(ap1 + 16 * hh);
    a1.h[1] = load_h8(ap1 + 16 * hh + 8);
#pragma unroll
    for (int t = 0; t < 4; ++t) {
      v16h b = load_h16(w16 + (n0 + t * 16 + l15) * DIM + kc + 16 * hh);
      acc[0][t] = wmma16(a0.v, b, acc[0][t]);
      acc[1][t] = wmma16(a1.v, b, acc[1][t]);
    }
  }

#pragma unroll
  for (int t = 0; t < 4; ++t) {
    const int col = n0 + t * 16 + l15;
    const float bv = bias[col];
#pragma unroll
    for (int u = 0; u < 2; ++u) {
#pragma unroll
      for (int r = 0; r < 8; ++r) {
        const int row = row0 + u * 16 + r + 8 * hh;
        out[(size_t)row * DIM + col] = acc[u][t][r] + bv;
      }
    }
  }
}

// ---------------------------------------------------------------- launch
extern "C" void kernel_launch(void* const* d_in, const int* in_sizes, int n_in,
                              void* d_out, int out_size, void* d_ws, size_t ws_size,
                              hipStream_t stream) {
  const float* x      = (const float*)d_in[0];  // [16,1024,768]
  const float* qkv_w  = (const float*)d_in[1];  // [2304,768]
  const float* qkv_b  = (const float*)d_in[2];  // [2304]
  /* d_in[3] qkv_B: feedback-alignment buffer, unused in forward */
  const float* proj_w = (const float*)d_in[4];  // [768,768]
  const float* proj_b = (const float*)d_in[5];  // [768]
  /* d_in[6] proj_B: unused */
  float* out = (float*)d_out;                   // [16,1024,768] f32

  char* ws = (char*)d_ws;
  size_t off = 0;
  auto carve = [&](size_t bytes) -> _Float16* {
    _Float16* p = (_Float16*)(ws + off);
    off += (bytes + 255) & ~(size_t)255;
    return p;
  };
  const size_t nX  = (size_t)MROWS * DIM;                // 12,582,912
  const size_t nW1 = (size_t)QKVOUT * DIM;               // 1,769,472
  const size_t nW2 = (size_t)DIM * DIM;                  // 589,824
  const size_t nH  = (size_t)BATCH * NHEADS * SEQ * HD;  // 12,582,912

  _Float16* x16  = carve(nX * 2);
  _Float16* w1   = carve(nW1 * 2);
  _Float16* w2   = carve(nW2 * 2);
  _Float16* q16  = carve(nH * 2);
  _Float16* k16  = carve(nH * 2);
  _Float16* vT16 = carve(nH * 2);
  _Float16* att  = carve(nX * 2);

  cvt_f32_f16_kernel<<<(int)((nX + 255) / 256), 256, 0, stream>>>(x, x16, (int)nX);
  cvt_f32_f16_kernel<<<(int)((nW1 + 255) / 256), 256, 0, stream>>>(qkv_w, w1, (int)nW1);
  cvt_f32_f16_kernel<<<(int)((nW2 + 255) / 256), 256, 0, stream>>>(proj_w, w2, (int)nW2);

  // 2304/64 = 36 column tiles, 16384/(8*32) = 64 row blocks
  qkv_gemm_kernel<<<dim3(QKVOUT / 64, MROWS / 256), 256, 0, stream>>>(
      x16, w1, qkv_b, q16, k16, vT16);

  // 1024/256 = 4 query blocks, 16*12 = 192 (batch, head) pairs
  attn_kernel<<<dim3(SEQ / 256, BATCH * NHEADS), 256, 0, stream>>>(
      q16, k16, vT16, att);

  proj_gemm_kernel<<<dim3(DIM / 64, MROWS / 256), 256, 0, stream>>>(
      att, w2, proj_b, out);
}
